// SequenceToPair_3453153706645
// MI455X (gfx1250) — matrix operand order repaired and checked by reading, not compiled, not run on hardware
//
#include <hip/hip_runtime.h>
#include <hip/hip_bf16.h>

typedef __attribute__((ext_vector_type(8)))  _Float16 v8h;
typedef __attribute__((ext_vector_type(16))) _Float16 v16h;
typedef __attribute__((ext_vector_type(8)))  float    v8f;

#define LSEQ 1024   // L
#define CS   1024   // sequence_state_dim
#define CZ   128    // pairwise_state_dim
#define DQK  64     // 2*INNER

// ---------------------------------------------------------------------------
// Kernel 1: fused LayerNorm (over C_S) + Linear(C_S -> 64), output f16 q|k.
// One block per sequence row; 256 threads (8 wave32).
// ---------------------------------------------------------------------------
__global__ __launch_bounds__(256) void ln_proj_kernel(
    const float* __restrict__ ss, const float* __restrict__ lnw,
    const float* __restrict__ lnb, const float* __restrict__ pw,
    const float* __restrict__ pb, _Float16* __restrict__ qkh)
{
  __shared__ float s_row[CS];
  __shared__ float s_red[256];
  __shared__ float s_mu, s_rstd;

  const int row = blockIdx.x;
  const int tid = threadIdx.x;
  const float* x = ss + (size_t)row * CS;

  float vals[4];
  float lsum = 0.f, lsq = 0.f;
#pragma unroll
  for (int k = 0; k < 4; ++k) {
    float v = x[tid + 256 * k];
    vals[k] = v;
    lsum += v;
    lsq  += v * v;
  }

  // mean
  s_red[tid] = lsum;
  __syncthreads();
  for (int s = 128; s > 0; s >>= 1) {
    if (tid < s) s_red[tid] += s_red[tid + s];
    __syncthreads();
  }
  if (tid == 0) s_mu = s_red[0] * (1.f / CS);
  __syncthreads();

  // variance (E[x^2] - mu^2)
  s_red[tid] = lsq;
  __syncthreads();
  for (int s = 128; s > 0; s >>= 1) {
    if (tid < s) s_red[tid] += s_red[tid + s];
    __syncthreads();
  }
  if (tid == 0) {
    float mu  = s_mu;
    float var = s_red[0] * (1.f / CS) - mu * mu;
    s_rstd = rsqrtf(var + 1e-5f);
  }
  __syncthreads();

  const float mu = s_mu, rstd = s_rstd;
#pragma unroll
  for (int k = 0; k < 4; ++k) {
    int c = tid + 256 * k;
    s_row[c] = (vals[k] - mu) * rstd * lnw[c] + lnb[c];
  }
  __syncthreads();

  // projection: 64 outputs, each a 1024-long dot; 4 partial segments per output
  const int d   = tid & 63;
  const int seg = tid >> 6;
  const float* wrow = pw + (size_t)d * CS + seg * 256;
  const float* nrow = s_row + seg * 256;
  float acc = 0.f;
#pragma unroll 4
  for (int c = 0; c < 256; ++c) acc += nrow[c] * wrow[c];
  s_red[tid] = acc;
  __syncthreads();
  if (tid < 64) {
    float t = s_red[tid] + s_red[tid + 64] + s_red[tid + 128] + s_red[tid + 192]
            + pb[tid];
    qkh[(size_t)row * DQK + tid] = (_Float16)t;   // d<32: q, d>=32: k
  }
}

// ---------------------------------------------------------------------------
// Kernel 2: convert o_w [128 x 64] f32 -> f16 (row-major kept; B-fragment
// layout for v_wmma wants contiguous K runs of each o_w row, so no transpose).
// ---------------------------------------------------------------------------
__global__ void cvt_ow_kernel(const float* __restrict__ ow,
                              _Float16* __restrict__ owh)
{
  int i = blockIdx.x * blockDim.x + threadIdx.x;
  if (i < CZ * DQK) owh[i] = (_Float16)ow[i];
}

// ---------------------------------------------------------------------------
// Kernel 3: pairwise outer-product + o_proj via WMMA.
// Tile = 16 pairs (fixed i, 16 consecutive j) x 128 z. One wave32 per tile.
// A0 (prod) / A1 (diff) are 16x32 f16 fragments built from q[j],k[i].
// B fragments come straight out of the f16 o_w copy.
// out[(i*L+j)*CZ + z] = WMMA(A0,B0) + WMMA(A1,B1) + o_b[z]
// ---------------------------------------------------------------------------
__global__ __launch_bounds__(256) void pair_wmma_kernel(
    const _Float16* __restrict__ qkh, const _Float16* __restrict__ owh,
    const float* __restrict__ ob, float* __restrict__ out)
{
  const int lane = threadIdx.x & 31;
  const int wv   = threadIdx.x >> 5;
  const int tile = blockIdx.x * 8 + wv;

  const int i  = tile >> 6;          // row index (k side)
  const int j0 = (tile & 63) << 4;   // column block start (q side)
  const int m    = lane & 15;        // A row within tile / D column n
  const int half = lane >> 4;

  // 16-bit A layout (contiguous, matches B): lane(l<16): K = 0..15 ; lane(l>=16): K = 16..31
  const _Float16* qrow = qkh + (size_t)(j0 + m) * DQK;        // q = d 0..31
  const _Float16* krow = qkh + (size_t)i * DQK + 32;          // k = d 32..63
  v8h qa = *(const v8h*)(qrow + half * 16);
  v8h qb = *(const v8h*)(qrow + half * 16 + 8);
  v8h ka = *(const v8h*)(krow + half * 16);
  v8h kb = *(const v8h*)(krow + half * 16 + 8);

  v8h pl = qa * ka;                  // v_pk_mul_f16
  v8h ph = qb * kb;
  v8h dl = qa - ka;                  // v_pk_sub_f16
  v8h dh = qb - kb;
  v16h a0 = __builtin_shufflevector(pl, ph, 0,1,2,3,4,5,6,7,8,9,10,11,12,13,14,15);
  v16h a1 = __builtin_shufflevector(dl, dh, 0,1,2,3,4,5,6,7,8,9,10,11,12,13,14,15);

  float* obase = out + ((size_t)i * LSEQ + j0) * CZ + m;

#pragma unroll
  for (int zb = 0; zb < 8; ++zb) {
    const int z = zb * 16 + m;       // this lane's D column
    // B 32x16 f16 layout: lane holds o_w[z, 16*half .. 16*half+15]
    const _Float16* wrow = owh + (size_t)z * DQK + half * 16;
    v16h b0 = *(const v16h*)(wrow);        // K = d 0..31  (prod weights)
    v16h b1 = *(const v16h*)(wrow + 32);   // K = d 32..63 (diff weights)

    v8f c = {};
    c = __builtin_amdgcn_wmma_f32_16x16x32_f16(false, a0, false, b0,
                                               (short)0, c, false, false);
    c = __builtin_amdgcn_wmma_f32_16x16x32_f16(false, a1, false, b1,
                                               (short)0, c, false, false);
    const float bias = ob[z];

    // D layout: VGPR r -> pair row r + 8*half; stream out non-temporally
    // (512 MB output > 192 MB L2: don't evict the hot q/k/o_w tables)
#pragma unroll
    for (int r = 0; r < 8; ++r) {
      __builtin_nontemporal_store(c[r] + bias,
          obase + (size_t)(r + 8 * half) * CZ + zb * 16);
    }
  }
}

// ---------------------------------------------------------------------------
extern "C" void kernel_launch(void* const* d_in, const int* in_sizes, int n_in,
                              void* d_out, int out_size, void* d_ws, size_t ws_size,
                              hipStream_t stream) {
  const float* ss  = (const float*)d_in[0];  // [1,1024,1024]
  const float* lnw = (const float*)d_in[1];  // [1024]
  const float* lnb = (const float*)d_in[2];  // [1024]
  const float* pw  = (const float*)d_in[3];  // [64,1024]
  const float* pb  = (const float*)d_in[4];  // [64]
  const float* ow  = (const float*)d_in[5];  // [128,64]
  const float* ob  = (const float*)d_in[6];  // [128]
  float* out = (float*)d_out;                // [1,1024,1024,128]

  _Float16* qkh = (_Float16*)d_ws;                                   // 128 KB
  _Float16* owh = (_Float16*)((char*)d_ws + (size_t)LSEQ * DQK * 2); //  16 KB

  ln_proj_kernel<<<LSEQ, 256, 0, stream>>>(ss, lnw, lnb, pw, pb, qkh);
  cvt_ow_kernel<<<(CZ * DQK + 255) / 256, 256, 0, stream>>>(ow, owh);

  // 1024 i x 64 j-blocks = 65536 wave-tiles; 8 waves (256 thr) per block
  pair_wmma_kernel<<<(LSEQ * 64) / 8, 256, 0, stream>>>(qkh, owh, ob, out);
}
